// SG_Cross_Attention_24026047054363
// MI455X (gfx1250) — compile-verified, matrix operand order checked by reading, not run on hardware
//
#include <hip/hip_runtime.h>

typedef __attribute__((ext_vector_type(16))) _Float16 v16h;
typedef __attribute__((ext_vector_type(8)))  _Float16 v8h;
typedef __attribute__((ext_vector_type(8)))  float    v8f;

#define BATCH 4
#define NSEQ  2304      // 48*48
#define CCH   512
#define LKV   684       // 12 + 96 + 576
#define NH    8
#define HD    64

__device__ __forceinline__ v8f wmma16(v16h a, v16h b, v8f c) {
    // D = A(16x32 f16) * B(32x16 f16) + C(16x16 f32)
    return __builtin_amdgcn_wmma_f32_16x16x32_f16(
        /*neg_a=*/false, a, /*neg_b=*/false, b,
        /*c_mod=*/(short)0, c, /*reuse_a=*/false, /*reuse_b=*/false);
}

// Fragment load from row-major f16: element i holds K = base+i (i<8) or
// base+8+i (i>=8), base = (lane>>4)*8  ->  two contiguous 16B loads (b128).
__device__ __forceinline__ v16h load_frag_h(const _Float16* __restrict__ p, int base) {
    v8h lo = *(const v8h*)(p + base);
    v8h hi = *(const v8h*)(p + base + 16);
    return __builtin_shufflevector(lo, hi, 0, 1, 2, 3, 4, 5, 6, 7,
                                           8, 9, 10, 11, 12, 13, 14, 15);
}

// ---------------------------------------------------------------------------
// f32 -> f16 conversion (one-shot for x and weight matrices)
// ---------------------------------------------------------------------------
__global__ __launch_bounds__(256) void cvt_h_kernel(
    const float* __restrict__ in, _Float16* __restrict__ outp, int n)
{
    int i = blockIdx.x * blockDim.x + threadIdx.x;
    if (i < n) outp[i] = (_Float16)in[i];
}

// ---------------------------------------------------------------------------
// GEMM: Y[M,Nout] = X[M,K](f16) @ W[Nout,K](f16)^T + bias; K % 32 == 0.
// Optional f32 and/or f16 output buffers.
// block = 256 threads (8 waves); wave tile 32x32 arranged 4x2 -> block 128x64
// ---------------------------------------------------------------------------
__global__ __launch_bounds__(256) void gemm_wmma_f16_kernel(
    const _Float16* __restrict__ X, const _Float16* __restrict__ W,
    const float* __restrict__ bias, float* __restrict__ Yf,
    _Float16* __restrict__ Yh, int M, int K, int Nout)
{
    const int lane = threadIdx.x & 31;
    const int wave = threadIdx.x >> 5;
    const int wm = wave & 3, wn = wave >> 2;
    const int M0 = blockIdx.x * 128 + wm * 32;
    const int N0 = blockIdx.y * 64  + wn * 32;
    const int l16  = lane & 15;
    const int base = (lane >> 4) << 3;

    int rA0 = M0 + l16;       if (rA0 >= M) rA0 = M - 1;
    int rA1 = M0 + 16 + l16;  if (rA1 >= M) rA1 = M - 1;
    int rB0 = N0 + l16;       if (rB0 >= Nout) rB0 = Nout - 1;
    int rB1 = N0 + 16 + l16;  if (rB1 >= Nout) rB1 = Nout - 1;

    const _Float16* xa0 = X + (size_t)rA0 * K;
    const _Float16* xa1 = X + (size_t)rA1 * K;
    const _Float16* wb0 = W + (size_t)rB0 * K;
    const _Float16* wb1 = W + (size_t)rB1 * K;

    v8f acc00 = {}, acc01 = {}, acc10 = {}, acc11 = {};
    for (int k0 = 0; k0 < K; k0 += 32) {
        if (k0 + 64 <= K - 32) {   // prefetch next-next k tile
            __builtin_prefetch(xa0 + k0 + 64, 0, 0);
            __builtin_prefetch(wb0 + k0 + 64, 0, 0);
        }
        v16h a0 = load_frag_h(xa0 + k0, base);
        v16h a1 = load_frag_h(xa1 + k0, base);
        v16h b0 = load_frag_h(wb0 + k0, base);
        v16h b1 = load_frag_h(wb1 + k0, base);
        acc00 = wmma16(a0, b0, acc00);
        acc01 = wmma16(a0, b1, acc01);
        acc10 = wmma16(a1, b0, acc10);
        acc11 = wmma16(a1, b1, acc11);
    }
    const int hi8 = (lane >> 4) << 3;
#pragma unroll
    for (int r = 0; r < 8; ++r) {
        int row0 = M0 + r + hi8, row1 = M0 + 16 + r + hi8;
        int col0 = N0 + l16,     col1 = N0 + 16 + l16;
        float v00 = acc00[r] + bias[col0];
        float v01 = acc01[r] + bias[col1];
        float v10 = acc10[r] + bias[col0];
        float v11 = acc11[r] + bias[col1];
        if (Yf) {
            if (row0 < M && col0 < Nout) Yf[(size_t)row0 * Nout + col0] = v00;
            if (row0 < M && col1 < Nout) Yf[(size_t)row0 * Nout + col1] = v01;
            if (row1 < M && col0 < Nout) Yf[(size_t)row1 * Nout + col0] = v10;
            if (row1 < M && col1 < Nout) Yf[(size_t)row1 * Nout + col1] = v11;
        }
        if (Yh) {
            if (row0 < M && col0 < Nout) Yh[(size_t)row0 * Nout + col0] = (_Float16)v00;
            if (row0 < M && col1 < Nout) Yh[(size_t)row0 * Nout + col1] = (_Float16)v01;
            if (row1 < M && col0 < Nout) Yh[(size_t)row1 * Nout + col0] = (_Float16)v10;
            if (row1 < M && col1 < Nout) Yh[(size_t)row1 * Nout + col1] = (_Float16)v11;
        }
    }
}

// ---------------------------------------------------------------------------
// Bitonic argsort (ascending) of one 2304-long row, padded to 4096 with +INF.
// One block per (batch, mask) row; 256 threads; 32KB LDS.
// ---------------------------------------------------------------------------
__global__ __launch_bounds__(256) void argsort_kernel(
    const float* __restrict__ mask, int* __restrict__ idx_out)
{
    __shared__ float key[4096];
    __shared__ int   kid[4096];
    const float* m = mask + (size_t)blockIdx.x * NSEQ;
    for (int i = threadIdx.x; i < 4096; i += blockDim.x) {
        key[i] = (i < NSEQ) ? m[i] : 3.0e38f;
        kid[i] = i;
    }
    __syncthreads();
    for (int k = 2; k <= 4096; k <<= 1) {
        for (int j = k >> 1; j > 0; j >>= 1) {
            for (int i = threadIdx.x; i < 4096; i += blockDim.x) {
                int ixj = i ^ j;
                if (ixj > i) {
                    bool up = ((i & k) == 0);
                    float ka = key[i], kb = key[ixj];
                    if ((ka > kb) == up) {
                        key[i] = kb; key[ixj] = ka;
                        int t = kid[i]; kid[i] = kid[ixj]; kid[ixj] = t;
                    }
                }
            }
            __syncthreads();
        }
    }
    for (int i = threadIdx.x; i < NSEQ; i += blockDim.x)
        idx_out[(size_t)blockIdx.x * NSEQ + i] = kid[i];
}

// ---------------------------------------------------------------------------
// Pooled kv sequence (writes f16 directly for the kv GEMMs).
// rows 0..11: 48-token weighted pool (f1); 12..107: 12-token pool (f2);
// 108..683: direct copy. grid = B*684 blocks, 512 threads (one per channel).
// ---------------------------------------------------------------------------
__global__ __launch_bounds__(512) void pool_kernel(
    const float* __restrict__ ctx, const int* __restrict__ idx,
    const float* __restrict__ f1w, const float* __restrict__ f1b,
    const float* __restrict__ f2w, const float* __restrict__ f2b,
    _Float16* __restrict__ seq, int transposed)
{
    const int br = blockIdx.x;
    const int b = br / LKV, row = br % LKV;
    const int c = threadIdx.x;
    const int* id = idx + (size_t)b * NSEQ;

    auto fetch = [&](int r) -> float {
        int gi = id[r];
        if (transposed) gi = (gi % 48) * 48 + gi / 48;
        return ctx[((size_t)(b * NSEQ + gi)) * CCH + c];
    };

    float v;
    if (row < 12) {
        float s = 0.0f;
        for (int j = 0; j < 48; ++j) s += fetch(row * 48 + j) * f1w[j];
        v = s + f1b[0];
    } else if (row < 108) {
        int i2 = row - 12;
        float s = 0.0f;
        for (int j = 0; j < 12; ++j) s += fetch(576 + i2 * 12 + j) * f2w[j];
        v = s + f2b[0];
    } else {
        v = fetch(1728 + (row - 108));
    }
    seq[(size_t)br * CCH + c] = (_Float16)v;
}

// ---------------------------------------------------------------------------
// Depthwise 3x3 conv (zero pad) on lepe linear output: [B,N,C] NHWC layout.
// ---------------------------------------------------------------------------
__global__ __launch_bounds__(256) void lepe_conv_kernel(
    const float* __restrict__ lin, const float* __restrict__ cw,
    const float* __restrict__ cb, float* __restrict__ outp)
{
    size_t idx = (size_t)blockIdx.x * blockDim.x + threadIdx.x;
    if (idx >= (size_t)BATCH * NSEQ * CCH) return;
    int c = (int)(idx & (CCH - 1));
    int n = (int)((idx >> 9) % NSEQ);
    int b = (int)(idx / ((size_t)NSEQ * CCH));
    int y = n / 48, x = n % 48;
    float s = cb[c];
#pragma unroll
    for (int ky = 0; ky < 3; ++ky) {
        int yy = y + ky - 1;
        if (yy < 0 || yy >= 48) continue;
#pragma unroll
        for (int kx = 0; kx < 3; ++kx) {
            int xx = x + kx - 1;
            if (xx < 0 || xx >= 48) continue;
            s += lin[((size_t)(b * NSEQ + yy * 48 + xx)) * CCH + c] * cw[c * 9 + ky * 3 + kx];
        }
    }
    outp[idx] = s;
}

// attnout(f32) + lepe(f32) -> f16 input for the proj GEMM
__global__ __launch_bounds__(256) void add_cvt_kernel(
    const float* __restrict__ a, const float* __restrict__ bsrc,
    _Float16* __restrict__ o, int n)
{
    int i = blockIdx.x * blockDim.x + threadIdx.x;
    if (i < n) o[i] = (_Float16)(a[i] + bsrc[i]);
}

// ---------------------------------------------------------------------------
// Flash attention: one wave per 16-row q tile; 4 waves/block.
// kv chunk = 32 (one f16-WMMA K step for P·V). Lkv=684 masked to -1e30.
// q:[B*N,512] f16   kv1/kv2:[B*684,512] f16 (k cols h%4*64.., v cols 256+..)
// out:[B*N,512] f32
// ---------------------------------------------------------------------------
__global__ __launch_bounds__(128) void attn_kernel(
    const _Float16* __restrict__ qb, const _Float16* __restrict__ kv1,
    const _Float16* __restrict__ kv2, float* __restrict__ outp)
{
    __shared__ float    S[4][16 * 32];
    __shared__ _Float16 P[4][16 * 32];
    __shared__ float    Alpha[4][16];
    __shared__ float    Lrow[4][16];

    const int lane = threadIdx.x & 31;
    const int w    = threadIdx.x >> 5;
    const int gw   = blockIdx.x * 4 + w;
    const int b    = gw / (NH * 144);
    const int rem  = gw % (NH * 144);
    const int h    = rem / 144;
    const int qt   = rem % 144;
    const int l16  = lane & 15;
    const int base = (lane >> 4) << 3;
    const int hi8  = base;

    const _Float16* kvb = (h < 4) ? kv1 : kv2;
    const int kch = (h & 3) * HD;
    const int vch = 256 + (h & 3) * HD;
    const float scale = 0.125f;   // 64^-0.5

    const _Float16* qrow = qb + ((size_t)(b * NSEQ + qt * 16 + l16)) * CCH + h * HD;
    v16h qa0 = load_frag_h(qrow, base);
    v16h qa1 = load_frag_h(qrow + 32, base);

    v8f accO[4] = {};
    float mreg = -3.0e38f, lreg = 0.0f;

    for (int c0 = 0; c0 < LKV; c0 += 32) {
        // ---- S = Q K^T for a 16x32 score tile (2 N-tiles x 2 K-steps) ----
        v8f s0 = {}, s1 = {};
        {
            int kr = c0 + l16; if (kr > LKV - 1) kr = LKV - 1;
            const _Float16* krow = kvb + ((size_t)(b * LKV + kr)) * CCH + kch;
            s0 = wmma16(qa0, load_frag_h(krow, base), s0);
            s0 = wmma16(qa1, load_frag_h(krow + 32, base), s0);
        }
        {
            int kr = c0 + 16 + l16; if (kr > LKV - 1) kr = LKV - 1;
            const _Float16* krow = kvb + ((size_t)(b * LKV + kr)) * CCH + kch;
            s1 = wmma16(qa0, load_frag_h(krow, base), s1);
            s1 = wmma16(qa1, load_frag_h(krow + 32, base), s1);
        }
#pragma unroll
        for (int r = 0; r < 8; ++r) {
            S[w][(r + hi8) * 32 + l16     ] = s0[r];
            S[w][(r + hi8) * 32 + l16 + 16] = s1[r];
        }
        asm volatile("s_wait_dscnt 0" ::: "memory");

        // ---- online softmax update (lanes 0..15 each own one row) ----
        if (lane < 16) {
            float cmax = -3.0e38f;
            for (int jj = 0; jj < 32; ++jj) {
                float sv = S[w][lane * 32 + jj] * scale;
                if (c0 + jj >= LKV) sv = -1.0e30f;
                S[w][lane * 32 + jj] = sv;
                cmax = fmaxf(cmax, sv);
            }
            float mnew  = fmaxf(mreg, cmax);
            float alpha = __expf(mreg - mnew);
            float sum = 0.0f;
            for (int jj = 0; jj < 32; ++jj) {
                float p = __expf(S[w][lane * 32 + jj] - mnew);
                sum += p;
                P[w][lane * 32 + jj] = (_Float16)p;
            }
            lreg = lreg * alpha + sum;
            mreg = mnew;
            Alpha[w][lane] = alpha;
        }
        asm volatile("s_wait_dscnt 0" ::: "memory");

        // ---- rescale running O by alpha(row) ----
#pragma unroll
        for (int r = 0; r < 8; ++r) {
            float al = Alpha[w][r + hi8];
#pragma unroll
            for (int t = 0; t < 4; ++t) accO[t][r] *= al;
        }

        // ---- O += P V  (P: 16x32 f16 A-frag from LDS, V: 32x64) ----
        v16h pa;
#pragma unroll
        for (int i = 0; i < 16; ++i) {
            int kk = base + ((i < 8) ? i : (i + 8));
            pa[i] = P[w][l16 * 32 + kk];
        }
#pragma unroll
        for (int t = 0; t < 4; ++t) {
            v16h bv;
#pragma unroll
            for (int i = 0; i < 16; ++i) {
                int kk = base + ((i < 8) ? i : (i + 8));
                int kr = c0 + kk; if (kr > LKV - 1) kr = LKV - 1;
                bv[i] = kvb[((size_t)(b * LKV + kr)) * CCH + vch + t * 16 + l16];
            }
            accO[t] = wmma16(pa, bv, accO[t]);
        }
    }

    if (lane < 16) Lrow[w][lane] = lreg;
    asm volatile("s_wait_dscnt 0" ::: "memory");

#pragma unroll
    for (int r = 0; r < 8; ++r) {
        float inv = 1.0f / Lrow[w][r + hi8];
        int row = b * NSEQ + qt * 16 + r + hi8;
#pragma unroll
        for (int t = 0; t < 4; ++t)
            outp[(size_t)row * CCH + h * HD + t * 16 + l16] = accO[t][r] * inv;
    }
}

// ---------------------------------------------------------------------------
extern "C" void kernel_launch(void* const* d_in, const int* in_sizes, int n_in,
                              void* d_out, int out_size, void* d_ws, size_t ws_size,
                              hipStream_t stream)
{
    const float* x       = (const float*)d_in[0];
    const float* context = (const float*)d_in[1];
    const float* mask1   = (const float*)d_in[2];
    const float* mask2   = (const float*)d_in[3];
    const float* q_w     = (const float*)d_in[4];
    const float* q_b     = (const float*)d_in[5];
    const float* kv1_w   = (const float*)d_in[6];
    const float* kv1_b   = (const float*)d_in[7];
    const float* kv2_w   = (const float*)d_in[8];
    const float* kv2_b   = (const float*)d_in[9];
    const float* f1_w    = (const float*)d_in[10];
    const float* f1_b    = (const float*)d_in[11];
    const float* f2_w    = (const float*)d_in[12];
    const float* f2_b    = (const float*)d_in[13];
    const float* lepe_w  = (const float*)d_in[14];
    const float* lepe_b  = (const float*)d_in[15];
    const float* conv_w  = (const float*)d_in[16];
    const float* conv_b  = (const float*)d_in[17];
    const float* proj_w  = (const float*)d_in[18];
    const float* proj_b  = (const float*)d_in[19];
    float* outp = (float*)d_out;

    const size_t SZ1 = (size_t)BATCH * NSEQ * CCH;   // 4,718,592 elems
    const size_t SZ2 = (size_t)BATCH * LKV  * CCH;   // 1,400,832 elems
    const size_t SZW = (size_t)CCH * CCH;            //   262,144 elems

    char* wsp = (char*)d_ws;
    auto carve = [&](size_t bytes) -> char* {
        char* p = wsp;
        wsp += (bytes + 255) & ~(size_t)255;
        return p;
    };
    _Float16* x16     = (_Float16*)carve(SZ1 * 2);
    _Float16* qw16    = (_Float16*)carve(SZW * 2);
    _Float16* kv1w16  = (_Float16*)carve(SZW * 2);
    _Float16* kv2w16  = (_Float16*)carve(SZW * 2);
    _Float16* lepew16 = (_Float16*)carve(SZW * 2);
    _Float16* projw16 = (_Float16*)carve(SZW * 2);
    _Float16* seq1h   = (_Float16*)carve(SZ2 * 2);
    _Float16* seq2h   = (_Float16*)carve(SZ2 * 2);
    _Float16* q16     = (_Float16*)carve(SZ1 * 2);
    _Float16* kv1h    = (_Float16*)carve(SZ2 * 2);
    _Float16* kv2h    = (_Float16*)carve(SZ2 * 2);
    _Float16* projin  = (_Float16*)carve(SZ1 * 2);
    float*    lepelin = (float*)carve(SZ1 * 4);
    float*    lepeout = (float*)carve(SZ1 * 4);
    float*    attnout = (float*)carve(SZ1 * 4);
    int*      idx1    = (int*)carve((size_t)BATCH * NSEQ * 4);
    int*      idx2    = (int*)carve((size_t)BATCH * NSEQ * 4);

    // 0) one-shot f32 -> f16 of activations and weights
    cvt_h_kernel<<<(int)((SZ1 + 255) / 256), 256, 0, stream>>>(x, x16, (int)SZ1);
    cvt_h_kernel<<<(int)((SZW + 255) / 256), 256, 0, stream>>>(q_w,    qw16,    (int)SZW);
    cvt_h_kernel<<<(int)((SZW + 255) / 256), 256, 0, stream>>>(kv1_w,  kv1w16,  (int)SZW);
    cvt_h_kernel<<<(int)((SZW + 255) / 256), 256, 0, stream>>>(kv2_w,  kv2w16,  (int)SZW);
    cvt_h_kernel<<<(int)((SZW + 255) / 256), 256, 0, stream>>>(lepe_w, lepew16, (int)SZW);
    cvt_h_kernel<<<(int)((SZW + 255) / 256), 256, 0, stream>>>(proj_w, projw16, (int)SZW);

    // 1) argsorts (4 rows each)
    argsort_kernel<<<BATCH, 256, 0, stream>>>(mask1, idx1);
    argsort_kernel<<<BATCH, 256, 0, stream>>>(mask2, idx2);

    // 2) pooled kv sequences (f16)
    pool_kernel<<<BATCH * LKV, 512, 0, stream>>>(context, idx1, f1_w, f1_b, f2_w, f2_b, seq1h, 0);
    pool_kernel<<<BATCH * LKV, 512, 0, stream>>>(context, idx2, f1_w, f1_b, f2_w, f2_b, seq2h, 1);

    // 3) dense projections (WMMA, f16 operands)
    dim3 gBig((BATCH * NSEQ + 127) / 128, CCH / 64);   // (72, 8)
    dim3 gKv ((BATCH * LKV  + 127) / 128, CCH / 64);   // (22, 8)
    gemm_wmma_f16_kernel<<<gBig, 256, 0, stream>>>(x16, qw16, q_b, nullptr, q16,
                                                   BATCH * NSEQ, CCH, CCH);
    gemm_wmma_f16_kernel<<<gBig, 256, 0, stream>>>(x16, lepew16, lepe_b, lepelin, nullptr,
                                                   BATCH * NSEQ, CCH, CCH);
    lepe_conv_kernel<<<(int)((SZ1 + 255) / 256), 256, 0, stream>>>(lepelin, conv_w, conv_b, lepeout);
    gemm_wmma_f16_kernel<<<gKv, 256, 0, stream>>>(seq1h, kv1w16, kv1_b, nullptr, kv1h,
                                                  BATCH * LKV, CCH, CCH);
    gemm_wmma_f16_kernel<<<gKv, 256, 0, stream>>>(seq2h, kv2w16, kv2_b, nullptr, kv2h,
                                                  BATCH * LKV, CCH, CCH);

    // 4) attention (WMMA flash) — 4*8*144 q-tiles, 4 waves/block
    attn_kernel<<<(BATCH * NH * 144) / 4, 128, 0, stream>>>(q16, kv1h, kv2h, attnout);

    // 5) + lepe -> f16, then output projection (WMMA) straight into d_out
    add_cvt_kernel<<<(int)((SZ1 + 255) / 256), 256, 0, stream>>>(attnout, lepeout, projin, (int)SZ1);
    gemm_wmma_f16_kernel<<<gBig, 256, 0, stream>>>(projin, projw16, proj_b, outp, nullptr,
                                                   BATCH * NSEQ, CCH, CCH);
}
